// CrossBlock_21114059227279
// MI455X (gfx1250) — compile-verified, matrix operand order checked by reading, not run on hardware
//
#include <hip/hip_runtime.h>
#include <hip/hip_bf16.h>
#include <stdint.h>

// ---------------------------------------------------------------------------
// CDNA5 (gfx1250) CrossBlock: bf16 WMMA GEMMs + dual-direction flash attention
// + ASYNCcnt-tracked global->LDS tile staging where no transpose is needed.
// ---------------------------------------------------------------------------

typedef __bf16 bf16_t;
typedef __attribute__((ext_vector_type(16))) __bf16 v16bf;
typedef __attribute__((ext_vector_type(8)))  float  v8f;
typedef __attribute__((ext_vector_type(4)))  int    v4i;

union Frag16 { v16bf v; uint32_t u[8]; };
union FragF  { v8f   v; float    f[8]; };

__device__ __forceinline__ bf16_t f2bf(float f) {
  uint32_t u = __builtin_bit_cast(uint32_t, f);
  uint32_t r = (u + 0x7FFFu + ((u >> 16) & 1u)) >> 16;   // round-to-nearest-even
  unsigned short s = (unsigned short)r;
  return __builtin_bit_cast(bf16_t, s);
}

// ---------------- async global->LDS (CDNA5 GLOBAL_LOAD_ASYNC_TO_LDS) -------
#if defined(__has_builtin)
#if __has_builtin(__builtin_amdgcn_global_load_async_to_lds_b128)
#define HAVE_ASYNC_LDS 1
#else
#define HAVE_ASYNC_LDS 0
#endif
#else
#define HAVE_ASYNC_LDS 0
#endif

typedef __attribute__((address_space(1))) v4i* gas_v4i_t;
typedef __attribute__((address_space(3))) v4i* las_v4i_t;

__device__ __forceinline__ void copy_b128_to_lds(const void* g, void* l) {
#if HAVE_ASYNC_LDS
  __builtin_amdgcn_global_load_async_to_lds_b128((gas_v4i_t)g, (las_v4i_t)l, 0, 0);
#else
  *(uint4*)l = *(const uint4*)g;
#endif
}

__device__ __forceinline__ void async_lds_wait() {
#if HAVE_ASYNC_LDS
#if __has_builtin(__builtin_amdgcn_s_wait_asynccnt)
  __builtin_amdgcn_s_wait_asynccnt(0);
#else
  asm volatile("s_wait_asynccnt 0x0" ::: "memory");
#endif
#endif
}

// ------------------------- f32 -> bf16 (strided) ---------------------------
__global__ void cvt_f32_bf16(const float* __restrict__ in, bf16_t* __restrict__ out,
                             int total, int inner, int ostride) {
  int i = blockIdx.x * blockDim.x + threadIdx.x;
  if (i >= total) return;
  int r = i / inner, c = i - r * inner;
  out[(size_t)r * ostride + c] = f2bf(in[i]);
}

// ----------------------------- generic GEMM --------------------------------
// C[M=grid.x*128, N=grid.y*64] = (A[M,K]bf16 @ B[K,N]bf16 + bias)*scale (+res)
// 8 waves: wave w owns rows [w*16, w*16+16) of the block tile, all 64 cols.
__global__ __launch_bounds__(256) void gemm_wmma(
    const bf16_t* __restrict__ A, int lda,
    const bf16_t* __restrict__ Bm, int ldb,
    const float* __restrict__ bias, float scale,
    bf16_t* __restrict__ outB, float* __restrict__ outF,
    const float* __restrict__ res, int ldo, int ldr, int K) {
  __shared__ __align__(16) bf16_t As[128 * 32];
  __shared__ __align__(16) bf16_t Bs[64 * 32];   // stored [n][k] (K-contiguous)

  const int bm = blockIdx.x * 128;
  const int bn = blockIdx.y * 64;
  const int t = threadIdx.x;
  const int wave = t >> 5;
  const int lane = t & 31;
  const int hf = lane >> 4;       // which 16-lane half of the wave
  const int ln = lane & 15;

  FragF acc[4];
#pragma unroll
  for (int i = 0; i < 4; ++i) acc[i].v = (v8f){0, 0, 0, 0, 0, 0, 0, 0};

  for (int k0 = 0; k0 < K; k0 += 32) {
    __syncthreads();
    {  // stage A tile 128x32 (async DMA to LDS; two b128 per thread)
      const int row = t >> 1;
      const int hs = (t & 1) * 16;
      const bf16_t* gsrc = A + (size_t)(bm + row) * lda + k0 + hs;
      bf16_t* ldst = As + row * 32 + hs;
      copy_b128_to_lds(gsrc, ldst);
      copy_b128_to_lds(gsrc + 8, ldst + 8);
      if (k0 + 32 < K) __builtin_prefetch(gsrc + 32, 0, 0);
    }
    {  // stage B tile 32x64 -> LDS transposed [n][k] (register path)
      const int kr = t >> 3;
      const int nc = (t & 7) * 8;
      union { uint4 q; bf16_t e[8]; } ub;
      ub.q = *(const uint4*)(Bm + (size_t)(k0 + kr) * ldb + bn + nc);
#pragma unroll
      for (int e = 0; e < 8; ++e) Bs[(nc + e) * 32 + kr] = ub.e[e];
    }
    async_lds_wait();
    __syncthreads();

    Frag16 af;
#pragma unroll
    for (int v = 0; v < 8; ++v) {
      const int kp = (v >> 2) * 16 + hf * 8 + (v & 3) * 2;
      af.u[v] = *(const uint32_t*)(As + (wave * 16 + ln) * 32 + kp);
    }
#pragma unroll
    for (int ns = 0; ns < 4; ++ns) {
      Frag16 bfr;
#pragma unroll
      for (int v = 0; v < 8; ++v) {
        const int kp = (v >> 2) * 16 + hf * 8 + (v & 3) * 2;
        bfr.u[v] = *(const uint32_t*)(Bs + (ns * 16 + ln) * 32 + kp);
      }
      acc[ns].v = __builtin_amdgcn_wmma_f32_16x16x32_bf16(
          false, af.v, false, bfr.v, (short)0, acc[ns].v, false, false);
    }
  }

#pragma unroll
  for (int ns = 0; ns < 4; ++ns) {
#pragma unroll
    for (int r = 0; r < 8; ++r) {
      const int row = bm + wave * 16 + hf * 8 + r;   // C/D layout: m = r + 8*half
      const int col = bn + ns * 16 + ln;
      float vo = (acc[ns].f[r] + bias[col]) * scale;
      if (outB) {
        outB[(size_t)row * ldo + col] = f2bf(vo);
      } else {
        if (res) vo += res[(size_t)row * ldr + col];
        outF[(size_t)row * ldo + col] = vo;
      }
    }
  }
}

// ------------------- dual-direction flash attention ------------------------
// dir0: Q=qk0,K=qk1,V=v1, logits*=match[i,j]  -> m0
// dir1: Q=qk1,K=qk0,V=v0, logits*=match[j,i]  -> m1
// 4 waves/WG, each wave owns a 16-row Q tile; 32-wide KV chunks shared via LDS.
__global__ __launch_bounds__(128) void flash_attn(
    const bf16_t* __restrict__ qk0, const bf16_t* __restrict__ qk1,
    const bf16_t* __restrict__ v0b, const bf16_t* __restrict__ v1b,
    const float* __restrict__ match,
    bf16_t* __restrict__ m0, bf16_t* __restrict__ m1) {
  const int NN = 2048, DD = 256, DH = 32;
  const int dir = blockIdx.z >> 1;
  const int b = blockIdx.z & 1;
  const int h = blockIdx.y;
  const int wave = threadIdx.x >> 5;
  const int lane = threadIdx.x & 31;
  const int hf = lane >> 4;
  const int ln = lane & 15;
  const int i0 = blockIdx.x * 64 + wave * 16;

  const bf16_t* Q  = dir ? qk1 : qk0;
  const bf16_t* Km = dir ? qk0 : qk1;
  const bf16_t* Vm = dir ? v0b : v1b;
  bf16_t* Out      = dir ? m1 : m0;
  const float* mb  = match + (size_t)b * NN * NN;

  __shared__ __align__(16) bf16_t Kt[32 * 32];        // [j][dh]
  __shared__ __align__(16) bf16_t Vt[32 * 32];        // [dh][j] (transposed)
  __shared__ __align__(16) bf16_t Pb[4][16 * 32];     // per-wave P scratch

  // Q fragment (16x32, A layout), straight from global
  Frag16 qf;
  {
    const size_t qbase = (size_t)(b * NN + i0 + ln) * DD + h * DH;
#pragma unroll
    for (int v = 0; v < 8; ++v) {
      const int kp = (v >> 2) * 16 + hf * 8 + (v & 3) * 2;
      qf.u[v] = *(const uint32_t*)(Q + qbase + kp);
    }
  }

  float mrow[8], lrow[8];
  FragF O0, O1;
#pragma unroll
  for (int r = 0; r < 8; ++r) { mrow[r] = -1e30f; lrow[r] = 0.f; }
  O0.v = (v8f){0, 0, 0, 0, 0, 0, 0, 0};
  O1.v = (v8f){0, 0, 0, 0, 0, 0, 0, 0};

  for (int j0 = 0; j0 < NN; j0 += 32) {
    __syncthreads();
    {  // stage K chunk row-major (async DMA), V chunk transposed (registers)
      const int jr = threadIdx.x >> 2;
      const int dh0 = (threadIdx.x & 3) * 8;
      const size_t gb = (size_t)(b * NN + j0 + jr) * DD + h * DH + dh0;
      copy_b128_to_lds(Km + gb, Kt + jr * 32 + dh0);
      union { uint4 q; bf16_t e[8]; } uv;
      uv.q = *(const uint4*)(Vm + gb);
#pragma unroll
      for (int e = 0; e < 8; ++e) Vt[(dh0 + e) * 32 + jr] = uv.e[e];
    }
    async_lds_wait();
    __syncthreads();

    // S = Q @ K^T for two 16-col tiles
    FragF S0, S1;
    {
      Frag16 kf;
#pragma unroll
      for (int v = 0; v < 8; ++v) {
        const int kp = (v >> 2) * 16 + hf * 8 + (v & 3) * 2;
        kf.u[v] = *(const uint32_t*)(Kt + ln * 32 + kp);
      }
      S0.v = __builtin_amdgcn_wmma_f32_16x16x32_bf16(
          false, qf.v, false, kf.v, (short)0, (v8f){0,0,0,0,0,0,0,0}, false, false);
#pragma unroll
      for (int v = 0; v < 8; ++v) {
        const int kp = (v >> 2) * 16 + hf * 8 + (v & 3) * 2;
        kf.u[v] = *(const uint32_t*)(Kt + (16 + ln) * 32 + kp);
      }
      S1.v = __builtin_amdgcn_wmma_f32_16x16x32_bf16(
          false, qf.v, false, kf.v, (short)0, (v8f){0,0,0,0,0,0,0,0}, false, false);
    }

    // match-weight + online softmax (rows live on VGPR r, cols across 16 lanes)
    float p0[8], p1[8];
#pragma unroll
    for (int r = 0; r < 8; ++r) {
      const int irow = i0 + hf * 8 + r;
      const int ja = j0 + ln, jb = j0 + 16 + ln;
      float s0 = S0.f[r] * (dir ? mb[(size_t)ja * NN + irow] : mb[(size_t)irow * NN + ja]);
      float s1 = S1.f[r] * (dir ? mb[(size_t)jb * NN + irow] : mb[(size_t)irow * NN + jb]);
      float cm = fmaxf(s0, s1);
#pragma unroll
      for (int off = 1; off < 16; off <<= 1) cm = fmaxf(cm, __shfl_xor(cm, off, 32));
      const float nm = fmaxf(mrow[r], cm);
      const float fac = __expf(mrow[r] - nm);
      const float e0 = __expf(s0 - nm), e1 = __expf(s1 - nm);
      float cs = e0 + e1;
#pragma unroll
      for (int off = 1; off < 16; off <<= 1) cs += __shfl_xor(cs, off, 32);
      lrow[r] = lrow[r] * fac + cs;
      mrow[r] = nm;
      O0.f[r] *= fac; O1.f[r] *= fac;
      p0[r] = e0; p1[r] = e1;
    }

    // C-layout P -> LDS -> A-layout fragment (within-wave round trip)
    bf16_t* pb = &Pb[wave][0];
#pragma unroll
    for (int r = 0; r < 8; ++r) {
      const int mr = hf * 8 + r;
      pb[mr * 32 + ln]      = f2bf(p0[r]);
      pb[mr * 32 + 16 + ln] = f2bf(p1[r]);
    }
    asm volatile("s_wait_dscnt 0" ::: "memory");
    Frag16 pf, vf0, vf1;
#pragma unroll
    for (int v = 0; v < 8; ++v) {
      const int kp = (v >> 2) * 16 + hf * 8 + (v & 3) * 2;
      pf.u[v]  = *(const uint32_t*)(pb + ln * 32 + kp);
      vf0.u[v] = *(const uint32_t*)(Vt + ln * 32 + kp);
      vf1.u[v] = *(const uint32_t*)(Vt + (16 + ln) * 32 + kp);
    }
    O0.v = __builtin_amdgcn_wmma_f32_16x16x32_bf16(
        false, pf.v, false, vf0.v, (short)0, O0.v, false, false);
    O1.v = __builtin_amdgcn_wmma_f32_16x16x32_bf16(
        false, pf.v, false, vf1.v, (short)0, O1.v, false, false);
  }

#pragma unroll
  for (int r = 0; r < 8; ++r) {
    const int row = i0 + hf * 8 + r;
    const float inv = 1.0f / lrow[r];
    const size_t ob = (size_t)(b * NN + row) * DD + h * DH;
    Out[ob + ln]      = f2bf(O0.f[r] * inv);
    Out[ob + 16 + ln] = f2bf(O1.f[r] * inv);
  }
}

// ------------------------ LayerNorm + exact GELU ---------------------------
__global__ __launch_bounds__(256) void ln_gelu(
    const float* __restrict__ hbuf, const float* __restrict__ gamma,
    const float* __restrict__ beta, bf16_t* __restrict__ gbuf) {
  const int row = blockIdx.x;
  const float* hp = hbuf + (size_t)row * 512;
  const int c0 = threadIdx.x, c1 = threadIdx.x + 256;
  float x0 = hp[c0], x1 = hp[c1];

  __shared__ float red[20];
  float s = x0 + x1, q = x0 * x0 + x1 * x1;
#pragma unroll
  for (int off = 16; off; off >>= 1) { s += __shfl_xor(s, off, 32); q += __shfl_xor(q, off, 32); }
  const int wave = threadIdx.x >> 5, lane = threadIdx.x & 31;
  if (lane == 0) { red[wave] = s; red[wave + 8] = q; }
  __syncthreads();
  if (threadIdx.x == 0) {
    float S = 0.f, Q = 0.f;
    for (int w = 0; w < 8; ++w) { S += red[w]; Q += red[w + 8]; }
    const float mu = S * (1.f / 512.f);
    red[16] = mu;
    red[17] = Q * (1.f / 512.f) - mu * mu;
  }
  __syncthreads();
  const float mu = red[16];
  const float rs = rsqrtf(red[17] + 1e-5f);
  float y0 = (x0 - mu) * rs * gamma[c0] + beta[c0];
  float y1 = (x1 - mu) * rs * gamma[c1] + beta[c1];
  y0 = 0.5f * y0 * (1.f + erff(y0 * 0.70710678118654752f));
  y1 = 0.5f * y1 * (1.f + erff(y1 * 0.70710678118654752f));
  gbuf[(size_t)row * 512 + c0] = f2bf(y0);
  gbuf[(size_t)row * 512 + c1] = f2bf(y1);
}

// ------------------------------- host side ---------------------------------
extern "C" void kernel_launch(void* const* d_in, const int* in_sizes, int n_in,
                              void* d_out, int out_size, void* d_ws, size_t ws_size,
                              hipStream_t stream) {
  const int B = 2, N = 2048, D = 256, H = 8, M = B * N, D2 = 2 * D;
  (void)in_sizes; (void)n_in; (void)out_size; (void)ws_size; (void)H;

  const float* x0    = (const float*)d_in[0];
  const float* x1    = (const float*)d_in[1];
  const float* match = (const float*)d_in[2];
  const float* Wqk   = (const float*)d_in[3];
  const float* bqk   = (const float*)d_in[4];
  const float* Wv    = (const float*)d_in[5];
  const float* bv    = (const float*)d_in[6];
  const float* Wo    = (const float*)d_in[7];
  const float* bo    = (const float*)d_in[8];
  const float* W1    = (const float*)d_in[9];
  const float* b1    = (const float*)d_in[10];
  const float* gamma = (const float*)d_in[11];
  const float* beta  = (const float*)d_in[12];
  const float* W2    = (const float*)d_in[13];
  const float* b2    = (const float*)d_in[14];
  float* out = (float*)d_out;

  char* ws = (char*)d_ws;
  auto alloc = [&](size_t bytes) {
    char* p = ws; ws += (bytes + 255) & ~(size_t)255; return p;
  };
  bf16_t* WqkB = (bf16_t*)alloc((size_t)D * D * 2);
  bf16_t* WvB  = (bf16_t*)alloc((size_t)D * D * 2);
  bf16_t* WoB  = (bf16_t*)alloc((size_t)D * D * 2);
  bf16_t* W1B  = (bf16_t*)alloc((size_t)D2 * D2 * 2);
  bf16_t* W2B  = (bf16_t*)alloc((size_t)D2 * D * 2);
  bf16_t* qk0s = (bf16_t*)alloc((size_t)M * D * 2);
  bf16_t* qk1s = (bf16_t*)alloc((size_t)M * D * 2);
  bf16_t* v0bb = (bf16_t*)alloc((size_t)M * D * 2);
  bf16_t* v1bb = (bf16_t*)alloc((size_t)M * D * 2);
  bf16_t* m0b  = (bf16_t*)alloc((size_t)M * D * 2);
  bf16_t* m1b  = (bf16_t*)alloc((size_t)M * D * 2);
  bf16_t* cat0 = (bf16_t*)alloc((size_t)M * D2 * 2);
  bf16_t* cat1 = (bf16_t*)alloc((size_t)M * D2 * 2);
  float*  h0   = (float*)alloc((size_t)M * D2 * 4);
  float*  h1   = (float*)alloc((size_t)M * D2 * 4);
  bf16_t* g0 = qk0s;  // 4MB alias over qk0s+qk1s (dead after attention)
  bf16_t* g1 = v0bb;  // 4MB alias over v0bb+v1bb (dead after attention)

  auto cvt = [&](const float* in, bf16_t* o, int total, int inner, int ostride) {
    cvt_f32_bf16<<<dim3((total + 255) / 256), dim3(256), 0, stream>>>(in, o, total, inner, ostride);
  };
  // weights + x into concat buffers (first 256 cols of cat = x)
  cvt(Wqk, WqkB, D * D, D * D, D * D);
  cvt(Wv,  WvB,  D * D, D * D, D * D);
  cvt(Wo,  WoB,  D * D, D * D, D * D);
  cvt(W1,  W1B,  D2 * D2, D2 * D2, D2 * D2);
  cvt(W2,  W2B,  D2 * D, D2 * D, D2 * D);
  cvt(x0, cat0, M * D, D, D2);
  cvt(x1, cat1, M * D, D, D2);

  const float s = 0.42044820762685725f;  // (DH^-0.5)^0.5 = 32^-0.25

  auto gemmB = [&](const bf16_t* A, int lda, const bf16_t* Bm, int ldb,
                   const float* bias, float scale, bf16_t* oB, int ldo, int Nc, int K) {
    gemm_wmma<<<dim3(M / 128, Nc / 64), dim3(256), 0, stream>>>(
        A, lda, Bm, ldb, bias, scale, oB, nullptr, nullptr, ldo, 0, K);
  };
  auto gemmF = [&](const bf16_t* A, int lda, const bf16_t* Bm, int ldb,
                   const float* bias, float* oF, const float* res, int ldo, int ldr,
                   int Nc, int K) {
    gemm_wmma<<<dim3(M / 128, Nc / 64), dim3(256), 0, stream>>>(
        A, lda, Bm, ldb, bias, 1.0f, nullptr, oF, res, ldo, ldr, K);
  };

  // projections: qk = (x@Wqk + bqk)*s ; v = x@Wv + bv
  gemmB(cat0, D2, WqkB, D, bqk, s,    qk0s, D, D, D);
  gemmB(cat1, D2, WqkB, D, bqk, s,    qk1s, D, D, D);
  gemmB(cat0, D2, WvB,  D, bv,  1.0f, v0bb, D, D, D);
  gemmB(cat1, D2, WvB,  D, bv,  1.0f, v1bb, D, D, D);

  // flash attention, both directions
  flash_attn<<<dim3(N / 64, H, 2 * B), dim3(128), 0, stream>>>(
      qk0s, qk1s, v0bb, v1bb, match, m0b, m1b);

  // output projection, written directly into cat[:, 256:512]
  gemmB(m0b, D, WoB, D, bo, 1.0f, cat0 + D, D2, D, D);
  gemmB(m1b, D, WoB, D, bo, 1.0f, cat1 + D, D2, D, D);

  // FFN: h = cat @ W1 + b1 (f32) ; LN + exact GELU -> bf16
  gemmF(cat0, D2, W1B, D2, b1, h0, nullptr, D2, 0, D2, D2);
  gemmF(cat1, D2, W1B, D2, b1, h1, nullptr, D2, 0, D2, D2);
  ln_gelu<<<dim3(M), dim3(256), 0, stream>>>(h0, gamma, beta, g0);
  ln_gelu<<<dim3(M), dim3(256), 0, stream>>>(h1, gamma, beta, g1);

  // out = x + g @ W2 + b2
  gemmF(g0, D2, W2B, D, b2, out,               x0, D, D, D, D2);
  gemmF(g1, D2, W2B, D, b2, out + (size_t)M * D, x1, D, D, D, D2);
}
